// NeRF_TCNN_54065048322322
// MI455X (gfx1250) — matrix operand order repaired and checked by reading, not compiled, not run on hardware
//
#include <hip/hip_runtime.h>
#include <math.h>

typedef __attribute__((ext_vector_type(16))) _Float16 v16h;
typedef __attribute__((ext_vector_type(8)))  _Float16 v8h;
typedef __attribute__((ext_vector_type(8)))  float    v8f;

#define HASH_T      524288u      // 2^19 entries per level
#define HASH_MASK   (HASH_T - 1u)
#define NLEVELS     16
#define TILE_PTS    128          // points per block
#define BLOCK_THREADS 256        // 8 wave32 waves; each wave owns 16 rows

// f16 weight image layout in d_ws (halves):
#define WOFF_W1T 0        // [128][32]
#define WOFF_W2T 4096     // [32][128]
#define WOFF_C1T 8192     // [64][64]
#define WOFF_C2T 12288    // [64][64]
#define WOFF_C3T 16384    // [16][64]
#define WTOTAL   18432    // padded to 9 x 4KB async chunks (36,864 B)

__device__ __constant__ unsigned kRes[NLEVELS] = {
    16u, 27u, 45u, 74u, 123u, 204u, 338u, 562u,
    934u, 1553u, 2581u, 4290u, 7132u, 11857u, 19711u, 32768u
};
// res^3 <= 2^19 only for levels 0..3 -> dense, rest hashed
#define P1 2654435761u
#define P2 805459861u

// ---- WMMA fragment loaders (layouts per CDNA5 ISA 7.12.2, wave32) ----

__device__ __forceinline__ v16h ld_pair(const _Float16* p0, const _Float16* p1) {
    v8h lo = *(const v8h*)p0;
    v8h hi = *(const v8h*)p1;
    return __builtin_shufflevector(lo, hi, 0,1,2,3,4,5,6,7,8,9,10,11,12,13,14,15);
}

// A matrix 16x32 f16, row-major in LDS with row stride lda (halves).
__device__ __forceinline__ v16h load_a(const _Float16* A, int lda, int m0, int k0, int lane) {
    int row = m0 + (lane & 15);
    int kb  = (lane & 16) ? 8 : 0;
    const _Float16* base = A + row * lda + k0 + kb;
    return ld_pair(base, base + 16);
}

// B matrix 32x16 f16, stored transposed in LDS: Bt[n][k], row stride ldb (halves).
__device__ __forceinline__ v16h load_b(const _Float16* Bt, int ldb, int n0, int k0, int lane) {
    int col = n0 + (lane & 15);
    int kb  = (lane & 16) ? 8 : 0;
    const _Float16* base = Bt + col * ldb + k0 + kb;
    return ld_pair(base, base + 16);
}

// One fused layer: each wave computes rows [m0, m0+16) x all ND columns.
// A fragments hoisted into registers once (reused across all N tiles).
template<int KD, int ND, int RELU>
__device__ __forceinline__ void mlp_layer(const _Float16* A, int lda,
                                          const _Float16* Bt, int ldb,
                                          _Float16* C, int ldc,
                                          int m0, int lane) {
    constexpr int KT = KD / 32;
    v16h afrag[KT];
#pragma unroll
    for (int kt = 0; kt < KT; ++kt)
        afrag[kt] = load_a(A, lda, m0, kt * 32, lane);
#pragma unroll
    for (int n0 = 0; n0 < ND; n0 += 16) {
        v8f acc = {};
#pragma unroll
        for (int kt = 0; kt < KT; ++kt) {
            v16h b = load_b(Bt, ldb, n0, kt * 32, lane);
            acc = __builtin_amdgcn_wmma_f32_16x16x32_f16(
                false, afrag[kt], false, b, (short)0, acc, false, false);
        }
        int col   = n0 + (lane & 15);
        int rbase = m0 + ((lane & 16) ? 8 : 0);
#pragma unroll
        for (int r = 0; r < 8; ++r) {
            float v = acc[r];
            if (RELU) v = fmaxf(v, 0.0f);
            C[(rbase + r) * ldc + col] = (_Float16)v;
        }
    }
}

// ---- Prep: convert/transpose/pad all weights once per launch into d_ws ----
__global__ __launch_bounds__(256)
void prep_weights(const float* __restrict__ w1,   // [32][128]
                  const float* __restrict__ w2,   // [128][31]
                  const float* __restrict__ c1,   // [46][64]
                  const float* __restrict__ c2,   // [64][64]
                  const float* __restrict__ c3,   // [64][3]
                  _Float16* __restrict__ ws)
{
    const int tid = threadIdx.x;
    for (int i = tid; i < 32 * 128; i += 256) {            // w1[k][n] -> W1T[n][k]
        int k = i >> 7, n = i & 127;
        ws[WOFF_W1T + n * 32 + k] = (_Float16)w1[i];
    }
    for (int i = tid; i < 128 * 31; i += 256) {            // w2[k][n] -> W2T[n][k]
        int k = i / 31, n = i - k * 31;
        ws[WOFF_W2T + n * 128 + k] = (_Float16)w2[i];
    }
    for (int k = tid; k < 128; k += 256)                   // pad col 31
        ws[WOFF_W2T + 31 * 128 + k] = (_Float16)0.0f;
    for (int i = tid; i < 46 * 64; i += 256) {             // c1[k][n] -> C1T[n][k]
        int k = i >> 6, n = i & 63;
        ws[WOFF_C1T + n * 64 + k] = (_Float16)c1[i];
    }
    for (int i = tid; i < 64 * 18; i += 256) {             // pad K 46..63
        int n = i / 18, k = 46 + (i - n * 18);
        ws[WOFF_C1T + n * 64 + k] = (_Float16)0.0f;
    }
    for (int i = tid; i < 64 * 64; i += 256) {             // c2[k][n] -> C2T[n][k]
        int k = i >> 6, n = i & 63;
        ws[WOFF_C2T + n * 64 + k] = (_Float16)c2[i];
    }
    for (int i = tid; i < 64 * 3; i += 256) {              // c3[k][n] -> C3T[n][k]
        int k = i / 3, n = i - k * 3;
        ws[WOFF_C3T + n * 64 + k] = (_Float16)c3[i];
    }
    for (int i = tid; i < 13 * 64; i += 256) {             // pad N 3..15
        int n = 3 + (i >> 6), k = i & 63;
        ws[WOFF_C3T + n * 64 + k] = (_Float16)0.0f;
    }
    for (int i = tid; i < WTOTAL - 17408; i += 256)        // tail pad to 36,864 B
        ws[17408 + i] = (_Float16)0.0f;
}

__global__ __launch_bounds__(BLOCK_THREADS)
void nerf_tcnn_fused(const float* __restrict__ inp,       // [N][6]
                     const float* __restrict__ table,     // [16][2^19][2]
                     const _Float16* __restrict__ wimg,   // prepped weight image
                     float* __restrict__ out,             // [N][4]
                     int N)
{
    __shared__ _Float16 sWeights[WTOTAL];       // 36 KB, filled by async DMA
    __shared__ _Float16 sBufA[TILE_PTS * 128];  // 32 KB activations ping
    __shared__ _Float16 sBufB[TILE_PTS * 64];   // 16 KB activations pong
    __shared__ float    sPos[TILE_PTS * 6];     // staged inputs
    __shared__ float    sSigma[TILE_PTS];

    const int tid  = threadIdx.x;
    const int lane = tid & 31;
    const int wave = tid >> 5;
    const int m0   = wave * 16;
    const long long base = (long long)blockIdx.x * TILE_PTS;

    // ---- Kick off async weight image copy global->LDS (ASYNCcnt) ----
    // 36,864 B = 2304 x 16 B chunks = 9 issues x 256 lanes.
    {
        unsigned ldsBase = (unsigned)(uintptr_t)(&sWeights[0]);  // low 32b of generic = LDS offset
#pragma unroll
        for (int it = 0; it < 9; ++it) {
            unsigned chunk  = (unsigned)(it * BLOCK_THREADS + tid) * 16u;
            unsigned ldsOff = ldsBase + chunk;
            asm volatile("global_load_async_to_lds_b128 %0, %1, %2 offset:0"
                         :: "v"(ldsOff), "v"(chunk), "s"(wimg)
                         : "memory");
        }
    }

    // ---- Stage input points (overlaps with async weight DMA) ----
    for (int i = tid; i < TILE_PTS * 6; i += BLOCK_THREADS) {
        long long g = base * 6 + i;
        sPos[i] = (g < (long long)N * 6) ? inp[g] : 0.0f;
    }
    __syncthreads();   // sPos visible; async weight DMA still in flight

    // ---- Multires hash-grid encode -> g[TILE_PTS][32] f16 in sBufB (stride 32) ----
    for (int t = tid; t < TILE_PTS * NLEVELS; t += BLOCK_THREADS) {
        int p   = t & (TILE_PTS - 1);
        int lvl = t / TILE_PTS;
        unsigned resu = kRes[lvl];
        float    resf = (float)resu;
        unsigned p0[3]; float w[3];
#pragma unroll
        for (int d = 0; d < 3; ++d) {
            float v = sPos[p * 6 + d];
            v = fminf(fmaxf((v + 16.0f) * (1.0f / 32.0f), 0.0f), 1.0f) * (resf - 1.0f);
            float f = floorf(v);
            p0[d] = (unsigned)f;
            w[d]  = v - f;
        }
        const bool dense = (lvl < 4);
        float acc0 = 0.0f, acc1 = 0.0f;
        const float2* tbl = (const float2*)table + (size_t)lvl * HASH_T;
#pragma unroll
        for (int o = 0; o < 8; ++o) {
            unsigned ox = (o >> 2) & 1, oy = (o >> 1) & 1, oz = o & 1;
            unsigned cx = p0[0] + ox; cx = (cx < resu - 1u) ? cx : (resu - 1u);
            unsigned cy = p0[1] + oy; cy = (cy < resu - 1u) ? cy : (resu - 1u);
            unsigned cz = p0[2] + oz; cz = (cz < resu - 1u) ? cz : (resu - 1u);
            unsigned idx = dense
                ? ((cx + cy * resu + cz * resu * resu) & HASH_MASK)
                : ((cx ^ (cy * P2) ^ (cz * P1)) & HASH_MASK);
            float2 f = tbl[idx];
            float wc = (ox ? w[0] : 1.0f - w[0]) *
                       (oy ? w[1] : 1.0f - w[1]) *
                       (oz ? w[2] : 1.0f - w[2]);
            acc0 += wc * f.x;
            acc1 += wc * f.y;
        }
        sBufB[p * 32 + 2 * lvl]     = (_Float16)acc0;
        sBufB[p * 32 + 2 * lvl + 1] = (_Float16)acc1;
    }

    // ---- Weights must be resident before first WMMA layer ----
    asm volatile("s_wait_asynccnt 0" ::: "memory");
    __syncthreads();

    const _Float16* sW1T = &sWeights[WOFF_W1T];
    const _Float16* sW2T = &sWeights[WOFF_W2T];
    const _Float16* sC1T = &sWeights[WOFF_C1T];
    const _Float16* sC2T = &sWeights[WOFF_C2T];
    const _Float16* sC3T = &sWeights[WOFF_C3T];

    // ---- sigma net: h1 = relu(g @ w1)  [P,32]x[32,128] -> sBufA ----
    mlp_layer<32, 128, 1>(sBufB, 32, sW1T, 32, sBufA, 128, m0, lane);
    __syncthreads();
    // ---- h2 = h1 @ w2  [P,128]x[128,32] -> sBufB (stride 32) ----
    mlp_layer<128, 32, 0>(sBufA, 128, sW2T, 128, sBufB, 32, m0, lane);
    __syncthreads();

    // ---- sigma extract + SH(16) + geo(30) concat -> hc[P][64] in sBufA ----
    if (tid < TILE_PTS) {
        int p = tid;
        sSigma[p] = (float)sBufB[p * 32 + 0];
        float dx = sPos[p * 6 + 3], dy = sPos[p * 6 + 4], dz = sPos[p * 6 + 5];
        float x = ((dx + 1.0f) * 0.5f) * 2.0f - 1.0f;
        float y = ((dy + 1.0f) * 0.5f) * 2.0f - 1.0f;
        float z = ((dz + 1.0f) * 0.5f) * 2.0f - 1.0f;
        float x2 = x * x, y2 = y * y, z2 = z * z;
        float xy = x * y, yz = y * z, xz = x * z;
        _Float16* hc = &sBufA[p * 64];
        hc[0]  = (_Float16)(0.28209479177387814f);
        hc[1]  = (_Float16)(-0.48860251190291987f * y);
        hc[2]  = (_Float16)(0.48860251190291987f * z);
        hc[3]  = (_Float16)(-0.48860251190291987f * x);
        hc[4]  = (_Float16)(1.0925484305920792f * xy);
        hc[5]  = (_Float16)(-1.0925484305920792f * yz);
        hc[6]  = (_Float16)(0.94617469575756f * z2 - 0.31539156525252005f);
        hc[7]  = (_Float16)(-1.0925484305920792f * xz);
        hc[8]  = (_Float16)(0.5462742152960396f * (x2 - y2));
        hc[9]  = (_Float16)(0.5900435899266435f * y * (-3.0f * x2 + y2));
        hc[10] = (_Float16)(2.890611442640554f * xy * z);
        hc[11] = (_Float16)(0.4570457994644657f * y * (1.0f - 5.0f * z2));
        hc[12] = (_Float16)(0.3731763325901154f * z * (5.0f * z2 - 3.0f));
        hc[13] = (_Float16)(0.4570457994644657f * x * (1.0f - 5.0f * z2));
        hc[14] = (_Float16)(1.445305721320277f * z * (x2 - y2));
        hc[15] = (_Float16)(0.5900435899266435f * x * (-x2 + 3.0f * y2));
    }
    for (int i = tid; i < TILE_PTS * 30; i += BLOCK_THREADS) { // geo_feat = h2[:,1:31]
        int p = i / 30, j = i - p * 30;
        sBufA[p * 64 + 16 + j] = sBufB[p * 32 + 1 + j];
    }
    for (int i = tid; i < TILE_PTS * 18; i += BLOCK_THREADS) { // pad cols 46..63
        int p = i / 18, j = i - p * 18;
        sBufA[p * 64 + 46 + j] = (_Float16)0.0f;
    }
    __syncthreads();

    // ---- color net: relu(hc @ c1) -> sBufB; relu(@ c2) -> sBufA ----
    mlp_layer<64, 64, 1>(sBufA, 64, sC1T, 64, sBufB, 64, m0, lane);
    __syncthreads();
    mlp_layer<64, 64, 1>(sBufB, 64, sC2T, 64, sBufA, 64, m0, lane);
    __syncthreads();

    // ---- final: sigmoid(hid2 @ c3) -> out[:,0:3]; sigma -> out[:,3] ----
    {
        v16h a0 = load_a(sBufA, 64, m0, 0, lane);
        v16h a1 = load_a(sBufA, 64, m0, 32, lane);
        v16h b0 = load_b(sC3T, 64, 0, 0, lane);
        v16h b1 = load_b(sC3T, 64, 0, 32, lane);
        v8f acc = {};
        acc = __builtin_amdgcn_wmma_f32_16x16x32_f16(
            false, a0, false, b0, (short)0, acc, false, false);
        acc = __builtin_amdgcn_wmma_f32_16x16x32_f16(
            false, a1, false, b1, (short)0, acc, false, false);
        int col   = lane & 15;
        int rbase = m0 + ((lane & 16) ? 8 : 0);
        if (col < 3) {
#pragma unroll
            for (int r = 0; r < 8; ++r) {
                long long row = base + rbase + r;
                if (row < N) {
                    float v = acc[r];
                    out[row * 4 + col] = 1.0f / (1.0f + expf(-v));
                }
            }
        }
    }
    if (tid < TILE_PTS && base + tid < N)
        out[(base + tid) * 4 + 3] = sSigma[tid];
}

extern "C" void kernel_launch(void* const* d_in, const int* in_sizes, int n_in,
                              void* d_out, int out_size, void* d_ws, size_t ws_size,
                              hipStream_t stream) {
    (void)n_in; (void)out_size; (void)ws_size;
    const float* inp   = (const float*)d_in[0];
    const float* table = (const float*)d_in[1];
    const float* w1    = (const float*)d_in[2];
    const float* w2    = (const float*)d_in[3];
    const float* c1    = (const float*)d_in[4];
    const float* c2    = (const float*)d_in[5];
    const float* c3    = (const float*)d_in[6];
    float* out = (float*)d_out;
    _Float16* wimg = (_Float16*)d_ws;   // needs 36,864 B of scratch

    prep_weights<<<dim3(1), dim3(256), 0, stream>>>(w1, w2, c1, c2, c3, wimg);

    int N = in_sizes[0] / 6;
    int blocks = (N + TILE_PTS - 1) / TILE_PTS;
    nerf_tcnn_fused<<<dim3(blocks), dim3(BLOCK_THREADS), 0, stream>>>(
        inp, table, wimg, out, N);
}
